// NAM_44796508897982
// MI455X (gfx1250) — matrix operand order repaired and checked by reading, not compiled, not run on hardware
//
#include <hip/hip_runtime.h>

// Problem dims
#define NN 4096
#define FF 128
#define HH 256
#define OO 64

typedef __attribute__((ext_vector_type(16))) __bf16 v16bf;
typedef __attribute__((ext_vector_type(8)))  __bf16 v8bf;
typedef __attribute__((ext_vector_type(8)))  float  v8f;
typedef __attribute__((ext_vector_type(4)))  unsigned uint32x4;
typedef __attribute__((ext_vector_type(8)))  int      int32x8;
typedef __attribute__((ext_vector_type(4)))  int      int32x4;

constexpr int ROWS      = 128;  // batch rows per workgroup
constexpr int FCH       = 16;   // features per workgroup (FF / gridDim.y)
constexpr int LSTRIDE   = 264;  // bf16 elems per LDS row (256 + 8 pad; 528B rows, bank-spread)

// Dynamic LDS layout (byte offsets)
constexpr int OFF_XCOL = 0;                                   // 128*4 = 512
constexpr int OFF_H1   = 512;                                 // 128*264*2 = 67584
constexpr int OFF_H2   = OFF_H1 + ROWS * LSTRIDE * 2;
constexpr int OFF_W2T  = OFF_H2 + ROWS * LSTRIDE * 2;         // 256*264*2 = 135168
constexpr int OFF_W3T  = OFF_W2T + HH * LSTRIDE * 2;          // 64*264*2  = 33792
constexpr int SMEM_BYTES = OFF_W3T + OO * LSTRIDE * 2;        // 304640 B (~297.5 KB < 320 KB)

// Workspace layout (bf16 elems): pre-transposed weights
constexpr size_t W2T_ELEMS = (size_t)FF * HH * HH;            // 8388608
constexpr size_t W3T_ELEMS = (size_t)FF * HH * OO;            // 2097152

__device__ __forceinline__ __bf16 f2bf(float f) {
    union { float f; unsigned u; } v; v.f = f;
    unsigned r = (v.u + 0x7FFFu + ((v.u >> 16) & 1u)) >> 16;  // round-to-nearest-even
    union { unsigned short s; __bf16 b; } o; o.s = (unsigned short)r;
    return o.b;
}

__device__ __forceinline__ unsigned pk2(float a, float b) {
    union { __bf16 b; unsigned short s; } x, y; x.b = f2bf(a); y.b = f2bf(b);
    return (unsigned)x.s | ((unsigned)y.s << 16);
}

__device__ __forceinline__ v16bf combine16(v8bf lo, v8bf hi) {
    return __builtin_shufflevector(lo, hi, 0,1,2,3,4,5,6,7,8,9,10,11,12,13,14,15);
}

// Async global->LDS copy, 16 bytes per lane (GLOBAL_LOAD_ASYNC_TO_LDS_B128, ASYNCcnt).
__device__ __forceinline__ void async_g2l_b128(const __bf16* g, __bf16* l) {
    unsigned lds = (unsigned)(uintptr_t)l;               // low 32 bits of shared ptr = LDS addr
    unsigned long long ga = (unsigned long long)(uintptr_t)g;
    asm volatile("global_load_async_to_lds_b128 %0, %1, off"
                 :: "v"(lds), "v"(ga) : "memory");
}
__device__ __forceinline__ void wait_async0() {
    asm volatile("s_wait_asynccnt 0x0" ::: "memory");
}

// TDM: DMA a 2D bf16 tile [rows][256] (row-major, stride 256) from global into LDS,
// inserting 16B of padding after every 512B row => LDS row stride = 528B = LSTRIDE.
// D# per CDNA5 ISA. Issued once per workgroup (wave 0).
// This toolchain exposes the 6-arg builtin: (g0, g1, g2, g3, g_extra, cpol).
__device__ __forceinline__ void tdm_load_tile(const __bf16* gsrc, void* ldst, int rows) {
    unsigned long long ga = (unsigned long long)(uintptr_t)gsrc;
    unsigned lds = (unsigned)(uintptr_t)ldst;
    uint32x4 g0 = { 1u,                                       // count=1, user descriptor
                    lds,                                      // lds_addr
                    (unsigned)(ga & 0xFFFFFFFFu),             // global_addr[31:0]
                    (unsigned)((ga >> 32) & 0x01FFFFFFu)      // global_addr[56:32]
                        | (2u << 30) };                       // type = 2 ("image")
    int32x8 g1 = { (int)((1u << 16)                           // data_size = 1 (2 bytes)
                       | (1u << 20)                           // pad_enable
                       | (6u << 22)                           // pad_interval: 128 DWORDs (one row)
                       | (3u << 25)),                         // pad_amount: 4 DWORDs (16 B)
                   (int)(256u << 16),                         // tensor_dim0 = 256  (bits[79:48])
                   (int)(256u << 16),                         // tensor_dim1 = 256  (bits[111:80])
                   (int)(256u << 16),                         // tile_dim0  = 256  (bits[127:112])
                   (int)(unsigned)rows,                       // tile_dim1  = rows (bits[143:128])
                   256,                                       // tensor_dim0_stride = 256
                   0, 0 };
    int32x4 gz4 = { 0, 0, 0, 0 };                             // groups 2/3 unused (2D tile)
    int32x8 gz8 = { 0, 0, 0, 0, 0, 0, 0, 0 };
    __builtin_amdgcn_tensor_load_to_lds(g0, g1, gz4, gz4, gz8, 0);
}

// ---- prepass 1: out[n,o] = sum_f b3[f,o] (bias + deterministic init for atomics) ----
__global__ void nam_init_out(const float* __restrict__ b3, float* __restrict__ out) {
    int idx = blockIdx.x * 256 + threadIdx.x;
    int o = idx & (OO - 1);
    float s = 0.f;
    #pragma unroll 4
    for (int f = 0; f < FF; ++f) s += b3[f * OO + o];
    out[idx] = s;
}

// ---- prepass 2: bf16 + transpose weights into workspace: w2t[f][n][k], w3t[f][n][k] ----
__global__ void nam_cvt_weights(const float* __restrict__ w2, const float* __restrict__ w3,
                                __bf16* __restrict__ w2t, __bf16* __restrict__ w3t) {
    size_t idx = (size_t)blockIdx.x * 256 + threadIdx.x;
    if (idx < W2T_ELEMS) {
        int f = (int)(idx >> 16), n = (int)(idx >> 8) & 255, k = (int)idx & 255;
        w2t[idx] = f2bf(w2[((size_t)f << 16) + ((size_t)k << 8) + n]);
    } else if (idx < W2T_ELEMS + W3T_ELEMS) {
        size_t j = idx - W2T_ELEMS;
        int f = (int)(j >> 14), n = (int)(j >> 8) & 63, k = (int)j & 255;
        w3t[j] = f2bf(w3[(size_t)f * (HH * OO) + (size_t)k * OO + n]);
    }
}

// ---- main: fused 3-layer grouped MLP, bf16 WMMA, TDM + async staging ----
__global__ __launch_bounds__(256, 1)
void nam_mlp_fused(const float* __restrict__ x,  const float* __restrict__ w1,
                   const float* __restrict__ b1, const float* __restrict__ b2,
                   const __bf16* __restrict__ w2t, const __bf16* __restrict__ w3t,
                   float* __restrict__ out) {
    extern __shared__ char smem[];
    float*  xcol   = (float*)(smem + OFF_XCOL);
    __bf16* h1     = (__bf16*)(smem + OFF_H1);
    __bf16* h2     = (__bf16*)(smem + OFF_H2);
    __bf16* w2tile = (__bf16*)(smem + OFF_W2T);   // [n][k], stride LSTRIDE
    __bf16* w3tile = (__bf16*)(smem + OFF_W3T);   // [n][k], stride LSTRIDE

    const int t    = threadIdx.x;
    const int wave = t >> 5;            // wave w owns M-tile w (rows 16w..16w+15)
    const int lane = t & 31;
    const int r0   = blockIdx.x * ROWS;
    const int f0   = blockIdx.y * FCH;

    // WMMA fragment addressing (wave32 layouts per CDNA5 ISA 7.12.2)
    const int arow    = lane & 15;
    const int koff    = (lane >> 4) * 8;
    const int bn      = lane & 15;
    const int bkh     = (lane >> 4) * 8;
    const int crow_hi = (lane >> 4) * 8;

    // async staging geometry (w3 tile): 16B per lane per step
    const int st_n = t >> 5;
    const int st_k = (t & 31) * 8;

    const v8f vzero = {};
    v8f acc3[4];
    #pragma unroll
    for (int i = 0; i < 4; ++i) acc3[i] = vzero;

    #pragma unroll 1
    for (int ff = 0; ff < FCH; ++ff) {
        const int f = f0 + ff;

        __syncthreads();   // previous iteration's LDS readers done

        // ---- stage: x column (DS), w2 tile (TDM), w3 tile (async-to-LDS) ----
        if (t < ROWS) xcol[t] = x[(size_t)(r0 + t) * FF + f];
        if (wave == 0)
            tdm_load_tile(w2t + (size_t)f * (HH * HH), w2tile, HH);
        {
            const __bf16* g3 = w3t + (size_t)f * (HH * OO);
            #pragma unroll
            for (int j = 0; j < 8; ++j)
                async_g2l_b128(g3 + j * 2048 + t * 8,
                               w3tile + (j * 8 + st_n) * LSTRIDE + st_k);
        }
        wait_async0();
        if (wave == 0) __builtin_amdgcn_s_wait_tensorcnt(0);
        __syncthreads();   // tiles + xcol visible to all waves

        // ---- layer 1: h1 = relu(x*w1 + b1) -> bf16, packed 16B stores ----
        {
            const int   r  = t >> 1;
            const int   c0 = (t & 1) * 128;
            const float xv = xcol[r];
            const float* w1p = w1 + (size_t)f * HH + c0;
            const float* b1p = b1 + (size_t)f * HH + c0;
            __bf16* hp = h1 + r * LSTRIDE + c0;
            #pragma unroll 2
            for (int c = 0; c < 128; c += 8) {
                float v[8];
                #pragma unroll
                for (int i = 0; i < 8; ++i) {
                    float u = fmaf(xv, w1p[c + i], b1p[c + i]);
                    v[i] = u > 0.f ? u : 0.f;
                }
                uint4 pk = { pk2(v[0], v[1]), pk2(v[2], v[3]),
                             pk2(v[4], v[5]), pk2(v[6], v[7]) };
                *(uint4*)(hp + c) = pk;
            }
        }
        // h1 rows 16w..16w+15 written by wave w and read only by wave w -> no barrier.

        // ---- layer 2: 128x256x256 GEMM, barrier-free over K ----
        v8f c2[16];
        #pragma unroll
        for (int i = 0; i < 16; ++i) c2[i] = vzero;
        #pragma unroll 1
        for (int k = 0; k < 8; ++k) {
            const __bf16* ap = h1 + (wave * 16 + arow) * LSTRIDE + k * 32 + koff;
            v16bf a = combine16(*(const v8bf*)ap, *(const v8bf*)(ap + 16));
            #pragma unroll
            for (int nt = 0; nt < 16; ++nt) {
                const __bf16* bp = w2tile + (nt * 16 + bn) * LSTRIDE + k * 32 + bkh;
                v16bf b = combine16(*(const v8bf*)bp, *(const v8bf*)(bp + 16));
                c2[nt] = __builtin_amdgcn_wmma_f32_16x16x32_bf16(
                    false, a, false, b, (short)0, c2[nt], false, false);
            }
        }

        // ---- bias + relu -> h2 (own rows only) ----
        {
            const float* b2f = b2 + (size_t)f * HH;
            #pragma unroll
            for (int nt = 0; nt < 16; ++nt) {
                const int col  = nt * 16 + bn;
                const float bs = b2f[col];
                #pragma unroll
                for (int i = 0; i < 8; ++i) {
                    float v = c2[nt][i] + bs;
                    h2[(wave * 16 + i + crow_hi) * LSTRIDE + col] =
                        f2bf(v > 0.f ? v : 0.f);
                }
            }
        }

        // ---- layer 3: 128x64x256 GEMM, accumulate over f, barrier-free ----
        #pragma unroll 1
        for (int k = 0; k < 8; ++k) {
            const __bf16* ap = h2 + (wave * 16 + arow) * LSTRIDE + k * 32 + koff;
            v16bf a = combine16(*(const v8bf*)ap, *(const v8bf*)(ap + 16));
            #pragma unroll
            for (int nt = 0; nt < 4; ++nt) {
                const __bf16* bp = w3tile + (nt * 16 + bn) * LSTRIDE + k * 32 + bkh;
                v16bf b = combine16(*(const v8bf*)bp, *(const v8bf*)(bp + 16));
                acc3[nt] = __builtin_amdgcn_wmma_f32_16x16x32_bf16(
                    false, a, false, b, (short)0, acc3[nt], false, false);
            }
        }
    }

    // ---- native f32 atomic accumulate of the [128,64] partial ----
    #pragma unroll
    for (int nt = 0; nt < 4; ++nt) {
        const int col = nt * 16 + bn;
        #pragma unroll
        for (int i = 0; i < 8; ++i) {
            const int row = r0 + wave * 16 + i + crow_hi;
            unsafeAtomicAdd(&out[(size_t)row * OO + col], acc3[nt][i]);
        }
    }
}

extern "C" void kernel_launch(void* const* d_in, const int* in_sizes, int n_in,
                              void* d_out, int out_size, void* d_ws, size_t ws_size,
                              hipStream_t stream) {
    const float* x  = (const float*)d_in[0];
    const float* w1 = (const float*)d_in[1];
    const float* b1 = (const float*)d_in[2];
    const float* w2 = (const float*)d_in[3];
    const float* b2 = (const float*)d_in[4];
    const float* w3 = (const float*)d_in[5];
    const float* b3 = (const float*)d_in[6];
    float* out = (float*)d_out;

    __bf16* w2t = (__bf16*)d_ws;                 // 16 MB
    __bf16* w3t = w2t + W2T_ELEMS;               // +4 MB (needs ws >= 21 MB)

    // prepasses
    nam_init_out<<<(NN * OO) / 256, 256, 0, stream>>>(b3, out);
    size_t cvt_total = W2T_ELEMS + W3T_ELEMS;
    nam_cvt_weights<<<(unsigned)((cvt_total + 255) / 256), 256, 0, stream>>>(w2, w3, w2t, w3t);

    // fused main kernel
    dim3 grid(NN / ROWS, FF / FCH);
    nam_mlp_fused<<<grid, 256, SMEM_BYTES, stream>>>(x, w1, b1, b2, w2t, w3t, out);
}
